// HybridMoE_18167711662614
// MI455X (gfx1250) — matrix operand order repaired and checked by reading, not compiled, not run on hardware
//
#include <hip/hip_runtime.h>
#include <hip/hip_bf16.h>

#define HIDDEN 1024
#define INTER  4096
#define NEXP   8
#define BN     64      // block tile N (4 waves x 16)
#define BK     32      // k-step per WMMA
#define SKP    40      // padded LDS row stride (bf16 elems): 80B rows, 16B aligned

typedef __attribute__((ext_vector_type(16))) __bf16 v16bf;
typedef __attribute__((ext_vector_type(8)))  __bf16 v8bf;
typedef __attribute__((ext_vector_type(8)))  float  v8f;

union Frag { v8bf h[2]; v16bf f; };

__device__ __forceinline__ float geluf(float v) {
    return 0.5f * v * (1.0f + erff(v * 0.70710678118654752f));
}

// ---------------- x -> bf16 ----------------
__global__ void cvt_x_kernel(const float* __restrict__ x, __bf16* __restrict__ xb, int n) {
    int i = blockIdx.x * blockDim.x + threadIdx.x;
    if (i < n) xb[i] = (__bf16)x[i];
}

// ---------------- gating: logits + argmax + counts ----------------
__global__ void route_kernel(const float* __restrict__ x, const float* __restrict__ Wg,
                             int* __restrict__ idx, int* __restrict__ counts, int N) {
    int t = blockIdx.x * blockDim.x + threadIdx.x;
    if (t >= N) return;
    float acc[NEXP];
#pragma unroll
    for (int e = 0; e < NEXP; e++) acc[e] = 0.0f;
    const float* xr = x + (size_t)t * HIDDEN;
    for (int c = 0; c < HIDDEN; c++) {
        float xv = xr[c];
        const float* wr = Wg + (size_t)c * NEXP;
#pragma unroll
        for (int e = 0; e < NEXP; e++) acc[e] += xv * wr[e];
    }
    int best = 0; float bv = acc[0];
#pragma unroll
    for (int e = 1; e < NEXP; e++) if (acc[e] > bv) { bv = acc[e]; best = e; }
    idx[t] = best;
    atomicAdd(&counts[best], 1);
}

// ---------------- build padded offsets + M-tile table ----------------
__global__ void finalize_kernel(const int* __restrict__ counts, int* __restrict__ offsets,
                                int* __restrict__ tileGid, int* __restrict__ tileBase,
                                int* __restrict__ nTiles, int N) {
    if (threadIdx.x != 0 || blockIdx.x != 0) return;
    int t = 0;
    for (int i = 0; i < N / 32; i++) { tileGid[t] = 0; tileBase[t] = i * 32; t++; } // shared group
    int off = 0;
    for (int e = 0; e < NEXP; e++) {
        offsets[e] = off;
        int padded = (counts[e] + 31) & ~31;
        for (int c = 0; c < padded / 32; c++) { tileGid[t] = e + 1; tileBase[t] = N + off + c * 32; t++; }
        off += padded;
    }
    nTiles[0] = t;
}

__global__ void fill_rowtoken_kernel(int* __restrict__ rowToken, int N, int slotMax) {
    int s = blockIdx.x * blockDim.x + threadIdx.x;
    if (s < slotMax) rowToken[s] = (s < N) ? s : -1;
}

__global__ void scatter_kernel(const int* __restrict__ idx, const int* __restrict__ offsets,
                               int* __restrict__ cursor, int* __restrict__ rowToken, int N) {
    int t = blockIdx.x * blockDim.x + threadIdx.x;
    if (t >= N) return;
    int e = idx[t];
    int pos = atomicAdd(&cursor[e], 1);
    rowToken[N + offsets[e] + pos] = t;
}

// ---------------- FFN1: h = gelu(x@W1) * (x@W2), bf16 WMMA, LDS-staged B ----------------
// block = 256 thr = 8 waves, tile M=32 (2 wave rows) x N=64 (4 wave cols), K-step 32
__global__ __launch_bounds__(256) void ffn1_kernel(
        const __bf16* __restrict__ xb,
        const float* __restrict__ w1s, const float* __restrict__ w2s,
        const float* __restrict__ W1e, const float* __restrict__ W2e,
        __bf16* __restrict__ hbuf, const int* __restrict__ rowToken,
        const int* __restrict__ tileGid, const int* __restrict__ tileBase,
        const int* __restrict__ nTiles) {
    __shared__ __bf16 sB1[BN * SKP];
    __shared__ __bf16 sB2[BN * SKP];

    int bx = blockIdx.x;
    if (bx >= nTiles[0]) return;
    int gid = tileGid[bx];
    int sbase = tileBase[bx];
    const float* B1 = (gid == 0) ? w1s : (W1e + (size_t)(gid - 1) * HIDDEN * INTER);
    const float* B2 = (gid == 0) ? w2s : (W2e + (size_t)(gid - 1) * HIDDEN * INTER);

    int tid = threadIdx.x, lane = tid & 31, wave = tid >> 5;
    int waveM = wave & 1, waveN = wave >> 1;
    int colBlock = blockIdx.y * BN;
    int lh = lane & 15, lq = lane >> 4;

    // cooperative B-tile load assignment: 64 columns x 4 k-groups of 8
    int cn  = tid & 63;
    int ckg = tid >> 6;
    const float* g1 = B1 + colBlock + cn + (size_t)(ckg * 8) * INTER;
    const float* g2 = B2 + colBlock + cn + (size_t)(ckg * 8) * INTER;
    __bf16* s1p = &sB1[cn * SKP + ckg * 8];
    __bf16* s2p = &sB2[cn * SKP + ckg * 8];

    // A row (gathered token)
    int slotRow = sbase + waveM * 16 + lh;
    int tok = rowToken[slotRow];
    if (tok < 0) tok = 0;                   // pad rows compute garbage; never stored downstream
    const __bf16* arow = xb + (size_t)tok * HIDDEN + lq * 16;

    // B fragment read pointers (transposed LDS tile: [n][k])
    const __bf16* f1p = &sB1[(waveN * 16 + lh) * SKP + lq * 16];
    const __bf16* f2p = &sB2[(waveN * 16 + lh) * SKP + lq * 16];

    v8f acc1 = {}, acc2 = {};
    for (int k0 = 0; k0 < HIDDEN; k0 += BK) {
        __syncthreads();
        v8bf p1, p2;
#pragma unroll
        for (int j = 0; j < 8; j++) {
            p1[j] = (__bf16)g1[(size_t)(k0 + j) * INTER];
            p2[j] = (__bf16)g2[(size_t)(k0 + j) * INTER];
        }
        *(v8bf*)s1p = p1;
        *(v8bf*)s2p = p2;
        __syncthreads();

        Frag a, b1, b2;
        a.h[0]  = *(const v8bf*)(arow + k0);
        a.h[1]  = *(const v8bf*)(arow + k0 + 8);
        b1.h[0] = *(const v8bf*)(f1p);
        b1.h[1] = *(const v8bf*)(f1p + 8);
        b2.h[0] = *(const v8bf*)(f2p);
        b2.h[1] = *(const v8bf*)(f2p + 8);
        acc1 = __builtin_amdgcn_wmma_f32_16x16x32_bf16(false, a.f, false, b1.f, (short)0, acc1, false, false);
        acc2 = __builtin_amdgcn_wmma_f32_16x16x32_bf16(false, a.f, false, b2.f, (short)0, acc2, false, false);
    }
#pragma unroll
    for (int i = 0; i < 8; i++) {
        int m = lq * 8 + i;
        float h = geluf(acc1[i]) * acc2[i];
        int row = sbase + waveM * 16 + m;
        hbuf[(size_t)row * INTER + colBlock + waveN * 16 + lh] = (__bf16)h;
    }
}

// ---------------- FFN2: y = h @ W3, LDS-staged B, scatter-add into out ----------------
__global__ __launch_bounds__(256) void ffn2_kernel(
        const __bf16* __restrict__ hbuf,
        const float* __restrict__ w3s, const float* __restrict__ W3e,
        float* __restrict__ out, const int* __restrict__ rowToken,
        const int* __restrict__ tileGid, const int* __restrict__ tileBase,
        const int* __restrict__ nTiles) {
    __shared__ __bf16 sB[BN * SKP];

    int bx = blockIdx.x;
    if (bx >= nTiles[0]) return;
    int gid = tileGid[bx];
    int sbase = tileBase[bx];
    const float* B3 = (gid == 0) ? w3s : (W3e + (size_t)(gid - 1) * INTER * HIDDEN);

    int tid = threadIdx.x, lane = tid & 31, wave = tid >> 5;
    int waveM = wave & 1, waveN = wave >> 1;
    int colBlock = blockIdx.y * BN;
    int lh = lane & 15, lq = lane >> 4;

    int cn  = tid & 63;
    int ckg = tid >> 6;
    const float* g3 = B3 + colBlock + cn + (size_t)(ckg * 8) * HIDDEN;
    __bf16* s3p = &sB[cn * SKP + ckg * 8];

    const __bf16* arow = hbuf + (size_t)(sbase + waveM * 16 + lh) * INTER + lq * 16;
    const __bf16* f3p = &sB[(waveN * 16 + lh) * SKP + lq * 16];

    v8f acc = {};
    for (int k0 = 0; k0 < INTER; k0 += BK) {
        __syncthreads();
        v8bf p3;
#pragma unroll
        for (int j = 0; j < 8; j++) {
            p3[j] = (__bf16)g3[(size_t)(k0 + j) * HIDDEN];
        }
        *(v8bf*)s3p = p3;
        __syncthreads();

        Frag a, b;
        a.h[0] = *(const v8bf*)(arow + k0);
        a.h[1] = *(const v8bf*)(arow + k0 + 8);
        b.h[0] = *(const v8bf*)(f3p);
        b.h[1] = *(const v8bf*)(f3p + 8);
        acc = __builtin_amdgcn_wmma_f32_16x16x32_bf16(false, a.f, false, b.f, (short)0, acc, false, false);
    }
#pragma unroll
    for (int i = 0; i < 8; i++) {
        int m = lq * 8 + i;
        int row = sbase + waveM * 16 + m;
        int tok = rowToken[row];
        if (tok >= 0) atomicAdd(&out[(size_t)tok * HIDDEN + colBlock + waveN * 16 + lh], acc[i]);
    }
}

// ---------------- host launcher ----------------
extern "C" void kernel_launch(void* const* d_in, const int* in_sizes, int n_in,
                              void* d_out, int out_size, void* d_ws, size_t ws_size,
                              hipStream_t stream) {
    const float* x   = (const float*)d_in[0];
    const float* w1s = (const float*)d_in[1];
    const float* w2s = (const float*)d_in[2];
    const float* w3s = (const float*)d_in[3];
    const float* W1  = (const float*)d_in[4];
    const float* W2  = (const float*)d_in[5];
    const float* W3  = (const float*)d_in[6];
    const float* Wg  = (const float*)d_in[7];
    float* out = (float*)d_out;

    int N = in_sizes[0] / HIDDEN;          // 4096 tokens
    int slotMax = 2 * N + 32 * NEXP;       // shared slots + padded expert slots

    // carve workspace
    char* ws = (char*)d_ws;
    size_t off = 0;
    auto carve = [&](size_t bytes) -> char* {
        char* p = ws + off;
        off = (off + bytes + 255) & ~(size_t)255;
        return p;
    };
    __bf16* xb   = (__bf16*)carve((size_t)N * HIDDEN * 2);
    __bf16* hbuf = (__bf16*)carve((size_t)slotMax * INTER * 2);
    int* idx     = (int*)carve((size_t)N * 4);
    int* counts  = (int*)carve(2 * NEXP * 4);   // counts[0..7] then cursor[0..7]
    int* cursor  = counts + NEXP;
    int* offsets = (int*)carve(NEXP * 4);
    int* rowToken = (int*)carve((size_t)slotMax * 4);
    int maxTiles = (N / 32) * 2 + NEXP;         // shared tiles + worst-case expert tiles
    int* tileGid  = (int*)carve((size_t)maxTiles * 4);
    int* tileBase = (int*)carve((size_t)maxTiles * 4);
    int* nTiles   = (int*)carve(4);
    (void)ws_size; (void)n_in;

    (void)hipMemsetAsync(d_out, 0, (size_t)out_size * sizeof(float), stream);
    (void)hipMemsetAsync(counts, 0, 2 * NEXP * sizeof(int), stream);

    int tot = N * HIDDEN;
    cvt_x_kernel<<<(tot + 255) / 256, 256, 0, stream>>>(x, xb, tot);
    route_kernel<<<(N + 255) / 256, 256, 0, stream>>>(x, Wg, idx, counts, N);
    finalize_kernel<<<1, 32, 0, stream>>>(counts, offsets, tileGid, tileBase, nTiles, N);
    fill_rowtoken_kernel<<<(slotMax + 255) / 256, 256, 0, stream>>>(rowToken, N, slotMax);
    scatter_kernel<<<(N + 255) / 256, 256, 0, stream>>>(idx, offsets, cursor, rowToken, N);

    dim3 g1(maxTiles, INTER / BN);
    ffn1_kernel<<<g1, 256, 0, stream>>>(xb, w1s, w2s, W1, W2, hbuf, rowToken,
                                        tileGid, tileBase, nTiles);
    dim3 g2(maxTiles, HIDDEN / BN);
    ffn2_kernel<<<g2, 256, 0, stream>>>(hbuf, w3s, W3, out, rowToken,
                                        tileGid, tileBase, nTiles);
}
